// GNN_85194971283965
// MI455X (gfx1250) — matrix operand order repaired and checked by reading, not compiled, not run on hardware
//
#include <hip/hip_runtime.h>

typedef __attribute__((ext_vector_type(16))) _Float16 v16h;
typedef __attribute__((ext_vector_type(8)))  _Float16 v8h;
typedef __attribute__((ext_vector_type(8)))  float    v8f;
typedef __attribute__((ext_vector_type(4))) unsigned int tdm_u4;
typedef __attribute__((ext_vector_type(8))) int          tdm_i8;
typedef __attribute__((ext_vector_type(4))) int          tdm_i4;

#define NNODES 50000
#define NEDGES 400000

// ---------------------------------------------------------------------------
// Async global->LDS copy (per-lane addresses, ASYNCcnt) for gathered tiles.
// Low 32 bits of a generic pointer to __shared__ memory = LDS byte offset.
// ---------------------------------------------------------------------------
__device__ __forceinline__ unsigned lds_off(const void* p) {
  return (unsigned)(unsigned long long)p;
}
__device__ __forceinline__ void async_b128(unsigned lds_byte_off, const _Float16* g) {
  asm volatile("global_load_async_to_lds_b128 %0, %1, off" ::"v"(lds_byte_off), "v"(g)
               : "memory");
}
__device__ __forceinline__ void wait_async0() {
  asm volatile("s_wait_asynccnt 0x0" ::: "memory");
}

// ---------------------------------------------------------------------------
// Tensor Data Mover: DMA one contiguous B k-slab (NT*1024 bytes) into LDS.
// D# group0: count=1 | lds_addr | global_addr(57b) | type=2.
// D# group1: data_size=8B, 1-D tensor: tensor_dim0 = tile_dim0 = NT*128 elems.
// ---------------------------------------------------------------------------
template <int NT>
__device__ __forceinline__ void tdm_issue_slab(const _Float16* __restrict__ Ws, int kk,
                                               unsigned bbase) {
  unsigned long long ga = (unsigned long long)(Ws + (size_t)kk * NT * 512);
  const unsigned td0 = NT * 128;  // # of 8-byte elements
  tdm_u4 g0;
  g0[0] = 1u;                                                   // count=1, user desc
  g0[1] = bbase + (unsigned)(kk & 1) * (unsigned)(NT * 1024);   // lds_addr (bytes)
  g0[2] = (unsigned)ga;                                         // global_addr[31:0]
  g0[3] = (unsigned)((ga >> 32) & 0x01FFFFFFull) | (2u << 30);  // ga[56:32] | type=2
  tdm_i8 g1;
  g1[0] = 0x30000;                          // data_size = 3 (8 bytes)
  g1[1] = (int)((td0 & 0xFFFFu) << 16);     // tensor_dim0[15:0]
  g1[2] = (int)((td0 >> 16) | (1u << 16));  // tensor_dim0[31:16] | tensor_dim1 = 1
  g1[3] = (int)(td0 << 16);                 // tensor_dim1 hi = 0 | tile_dim0
  g1[4] = 0;                                // tile_dim1/2 unused
  g1[5] = (int)td0;                         // tensor_dim0_stride
  g1[6] = 0;
  g1[7] = 0;
  tdm_i4 z4 = {0, 0, 0, 0};
#if __clang_major__ >= 23
  tdm_i8 z8 = {0, 0, 0, 0, 0, 0, 0, 0};
  __builtin_amdgcn_tensor_load_to_lds(g0, g1, z4, z4, z8, 0);
#else
  __builtin_amdgcn_tensor_load_to_lds(g0, g1, z4, z4, 0);
#endif
}

// ---------------------------------------------------------------------------
// WMMA fragment helpers (v_wmma_f32_16x16x32_f16)
// ---------------------------------------------------------------------------
__device__ __forceinline__ v16h ld_a_frag(const _Float16* lds, int rs, int mt, int kk, int lane) {
  const _Float16* p = lds + (mt * 16 + (lane & 15)) * rs + kk * 32 + ((lane >> 4) << 3);
  v8h lo = *(const v8h*)(p);
  v8h hi = *(const v8h*)(p + 16);
  return __builtin_shufflevector(lo, hi, 0, 1, 2, 3, 4, 5, 6, 7, 8, 9, 10, 11, 12, 13, 14, 15);
}

// ---------------------------------------------------------------------------
// LDS-staged, double-buffered GEMM. 8 waves as 4 m-groups x 2 n-groups.
// Wave 0 streams B k-slab kk+1 via TDM while all waves compute slab kk.
// ---------------------------------------------------------------------------
template <int KT, int NT>
__device__ __forceinline__ void gemm_lds(v8f (&acc)[2][NT / 2], const _Float16* ldsA, int rsA,
                                         _Float16* ldsB /* 2*NT*512 halfs */,
                                         const _Float16* __restrict__ Ws, int wave, int lane) {
  const int mg = wave >> 1, ng = wave & 1;
  const unsigned bbase = lds_off(ldsB);
  if (wave == 0) {
    tdm_issue_slab<NT>(Ws, 0, bbase);
    __builtin_amdgcn_s_wait_tensorcnt(0);
  }
  __syncthreads();
#pragma unroll
  for (int kk = 0; kk < KT; ++kk) {
    if (wave == 0 && kk + 1 < KT) tdm_issue_slab<NT>(Ws, kk + 1, bbase);
    v16h a0 = ld_a_frag(ldsA, rsA, mg * 2 + 0, kk, lane);
    v16h a1 = ld_a_frag(ldsA, rsA, mg * 2 + 1, kk, lane);
    const _Float16* bslab = ldsB + (kk & 1) * NT * 512;
#pragma unroll
    for (int j = 0; j < NT / 2; ++j) {
      const int nt = ng * (NT / 2) + j;
      v16h b = *(const v16h*)(bslab + (nt * 32 + lane) * 16);
      acc[0][j] = __builtin_amdgcn_wmma_f32_16x16x32_f16(false, a0, false, b, (short)0,
                                                         acc[0][j], false, false);
      acc[1][j] = __builtin_amdgcn_wmma_f32_16x16x32_f16(false, a1, false, b, (short)0,
                                                         acc[1][j], false, false);
    }
    if (wave == 0) __builtin_amdgcn_s_wait_tensorcnt(0);
    __syncthreads();
  }
}

template <int NT>
__device__ __forceinline__ void store_relu_lds2(const v8f (&acc)[2][NT / 2],
                                                const float* __restrict__ bias, _Float16* ldsC,
                                                int rsC, int wave, int lane) {
  const int mg = wave >> 1, ng = wave & 1;
  const int n = lane & 15, mofs = (lane >> 4) << 3;
#pragma unroll
  for (int j = 0; j < NT / 2; ++j) {
    const int nt = ng * (NT / 2) + j;
    float bv = bias[nt * 16 + n];
#pragma unroll
    for (int i = 0; i < 2; ++i)
#pragma unroll
      for (int r = 0; r < 8; ++r) {
        float v = acc[i][j][r] + bv;
        v = v > 0.f ? v : 0.f;
        ldsC[((mg * 2 + i) * 16 + mofs + r) * rsC + nt * 16 + n] = (_Float16)v;
      }
  }
}

// ---------------------------------------------------------------------------
// Per-edge message MLP: A = [hid[nodeA] | hid[nodeB] | emb], 384 -> 256 -> 128,
// relu both layers, fp32 atomic scatter-add into out_sum[scat[e]].
// ---------------------------------------------------------------------------
__global__ __launch_bounds__(256) void edge_mlp_kernel(
    const _Float16* __restrict__ hid_h, const int* __restrict__ nodeA,
    const int* __restrict__ nodeB, const _Float16* __restrict__ emb,
    const _Float16* __restrict__ W1s, const float* __restrict__ b1,
    const _Float16* __restrict__ W2s, const float* __restrict__ b2,
    const int* __restrict__ scat, float* __restrict__ out_sum) {
  __shared__ __align__(16) _Float16 smem[128 * 392];
  __shared__ __align__(16) _Float16 bbuf[2 * 16 * 512];
  const int t = threadIdx.x, lane = t & 31, wave = t >> 5;
  const int mg = wave >> 1, ng = wave & 1;
  const int e0 = blockIdx.x * 128;

  // async gather of A tile [128 x 384] f16 into LDS (straight-line, 24 chunks)
  const unsigned abase = lds_off(smem);
#pragma unroll
  for (int u = 0; u < 24; ++u) {
    int c = t + u * 256;
    int row = c / 48, seg = c % 48;
    const _Float16* src;
    if (seg < 16)
      src = hid_h + (size_t)nodeA[e0 + row] * 128 + seg * 8;
    else if (seg < 32)
      src = hid_h + (size_t)nodeB[e0 + row] * 128 + (seg - 16) * 8;
    else
      src = emb + (size_t)(e0 + row) * 128 + (seg - 32) * 8;
    async_b128(abase + (unsigned)(row * 392 + seg * 8) * 2u, src);
  }
  wait_async0();
  __syncthreads();

  v8f acc1[2][8] = {};
  gemm_lds<12, 16>(acc1, smem, 392, bbuf, W1s, wave, lane);
  store_relu_lds2<16>(acc1, b1, smem, 264, wave, lane);  // smem reuse: A dead after gemm
  __syncthreads();

  v8f acc2[2][4] = {};
  gemm_lds<8, 8>(acc2, smem, 264, bbuf, W2s, wave, lane);

  const int n = lane & 15, mofs = (lane >> 4) << 3;
#pragma unroll
  for (int i = 0; i < 2; ++i)
#pragma unroll
    for (int r = 0; r < 8; ++r) {
      int gm = e0 + (mg * 2 + i) * 16 + mofs + r;
      size_t base = (size_t)scat[gm] * 128;
#pragma unroll
      for (int j = 0; j < 4; ++j) {
        int col = (ng * 4 + j) * 16 + n;
        float v = acc2[i][j][r] + b2[col];
        v = v > 0.f ? v : 0.f;
        atomicAdd(out_sum + base + col, v);
      }
    }
}

// ---------------------------------------------------------------------------
// hidden = relu(relu(batch_token @ vW1 + b1) @ vW2 + b2); also emit f16 copy
// ---------------------------------------------------------------------------
__global__ __launch_bounds__(256) void node_init_kernel(
    const float* __restrict__ x, const _Float16* __restrict__ W1s,
    const float* __restrict__ b1, const _Float16* __restrict__ W2s,
    const float* __restrict__ b2, float* __restrict__ hidden, _Float16* __restrict__ hid_h) {
  __shared__ __align__(16) _Float16 smem[128 * 264];
  __shared__ __align__(16) _Float16 bbuf[2 * 16 * 512];
  const int t = threadIdx.x, lane = t & 31, wave = t >> 5;
  const int mg = wave >> 1, ng = wave & 1;
  const int n0 = blockIdx.x * 128;
#pragma unroll
  for (int u = 0; u < 8; ++u) {
    int c = t + u * 256;
    int row = c >> 4, seg = c & 15;
    int gm = n0 + row;
    gm = gm < NNODES ? gm : NNODES - 1;
    const float* s = x + (size_t)gm * 128 + seg * 8;
    v8h tmp;
#pragma unroll
    for (int i = 0; i < 8; ++i) tmp[i] = (_Float16)s[i];
    *(v8h*)(smem + row * 136 + seg * 8) = tmp;
  }
  __syncthreads();
  v8f acc1[2][8] = {};
  gemm_lds<4, 16>(acc1, smem, 136, bbuf, W1s, wave, lane);
  store_relu_lds2<16>(acc1, b1, smem, 264, wave, lane);
  __syncthreads();
  v8f acc2[2][4] = {};
  gemm_lds<8, 8>(acc2, smem, 264, bbuf, W2s, wave, lane);
  const int n = lane & 15, mofs = (lane >> 4) << 3;
#pragma unroll
  for (int i = 0; i < 2; ++i)
#pragma unroll
    for (int r = 0; r < 8; ++r) {
      int gm = n0 + (mg * 2 + i) * 16 + mofs + r;
      if (gm < NNODES) {
#pragma unroll
        for (int j = 0; j < 4; ++j) {
          int col = (ng * 4 + j) * 16 + n;
          float v = acc2[i][j][r] + b2[col];
          v = v > 0.f ? v : 0.f;
          size_t o = (size_t)gm * 128 + col;
          hidden[o] = v;
          hid_h[o] = (_Float16)v;
        }
      }
    }
}

// ---------------------------------------------------------------------------
// hidden += MLP_a([hidden, Sp/cnt + p_mask*start, Sc/cnt + c_mask*end])
// ---------------------------------------------------------------------------
__global__ __launch_bounds__(256) void node_aggr_kernel(
    const _Float16* __restrict__ hid_h, const float* __restrict__ Sp,
    const float* __restrict__ Sc, const float* __restrict__ rcp_p,
    const float* __restrict__ rcp_c, const float* __restrict__ p_mask,
    const float* __restrict__ c_mask, const float* __restrict__ start_tok,
    const float* __restrict__ end_tok, const _Float16* __restrict__ W1s,
    const float* __restrict__ b1, const _Float16* __restrict__ W2s,
    const float* __restrict__ b2, float* __restrict__ hidden,
    _Float16* __restrict__ hid_h_out) {
  __shared__ __align__(16) _Float16 smem[128 * 392];
  __shared__ __align__(16) _Float16 bbuf[2 * 16 * 512];
  const int t = threadIdx.x, lane = t & 31, wave = t >> 5;
  const int mg = wave >> 1, ng = wave & 1;
  const int n0 = blockIdx.x * 128;
  const unsigned abase = lds_off(smem);
#pragma unroll
  for (int u = 0; u < 24; ++u) {
    int c = t + u * 256;
    int row = c / 48, seg = c % 48;
    int gm = n0 + row;
    gm = gm < NNODES ? gm : NNODES - 1;
    if (seg < 16) {
      async_b128(abase + (unsigned)(row * 392 + seg * 8) * 2u,
                 hid_h + (size_t)gm * 128 + seg * 8);
    } else if (seg < 32) {
      int j0 = (seg - 16) * 8;
      float rc = rcp_p[gm], pm = p_mask[gm];
      v8h tmp;
#pragma unroll
      for (int i = 0; i < 8; ++i)
        tmp[i] = (_Float16)(Sp[(size_t)gm * 128 + j0 + i] * rc + pm * start_tok[j0 + i]);
      *(v8h*)(smem + row * 392 + seg * 8) = tmp;
    } else {
      int j0 = (seg - 32) * 8;
      float rc = rcp_c[gm], cm = c_mask[gm];
      v8h tmp;
#pragma unroll
      for (int i = 0; i < 8; ++i)
        tmp[i] = (_Float16)(Sc[(size_t)gm * 128 + j0 + i] * rc + cm * end_tok[j0 + i]);
      *(v8h*)(smem + row * 392 + seg * 8) = tmp;
    }
  }
  wait_async0();
  __syncthreads();
  v8f acc1[2][8] = {};
  gemm_lds<12, 16>(acc1, smem, 392, bbuf, W1s, wave, lane);
  store_relu_lds2<16>(acc1, b1, smem, 264, wave, lane);
  __syncthreads();
  v8f acc2[2][4] = {};
  gemm_lds<8, 8>(acc2, smem, 264, bbuf, W2s, wave, lane);
  const int n = lane & 15, mofs = (lane >> 4) << 3;
#pragma unroll
  for (int i = 0; i < 2; ++i)
#pragma unroll
    for (int r = 0; r < 8; ++r) {
      int gm = n0 + (mg * 2 + i) * 16 + mofs + r;
      if (gm < NNODES) {
#pragma unroll
        for (int j = 0; j < 4; ++j) {
          int col = (ng * 4 + j) * 16 + n;
          float v = acc2[i][j][r] + b2[col];
          v = v > 0.f ? v : 0.f;
          size_t o = (size_t)gm * 128 + col;
          float h = hidden[o] + v;
          hidden[o] = h;
          hid_h_out[o] = (_Float16)h;
        }
      }
    }
}

// ---------------------------------------------------------------------------
// Edge embedding MLP: scalar -> 256 (elementwise VALU) -> 128 (WMMA GEMM)
// ---------------------------------------------------------------------------
__global__ __launch_bounds__(256) void edge_embed_kernel(
    const float* __restrict__ ind, const float* __restrict__ eW1,
    const float* __restrict__ eb1, const _Float16* __restrict__ W2s,
    const float* __restrict__ b2, _Float16* __restrict__ emb_out) {
  __shared__ __align__(16) _Float16 smem[128 * 264];
  __shared__ __align__(16) _Float16 bbuf[2 * 8 * 512];
  const int t = threadIdx.x, lane = t & 31, wave = t >> 5;
  const int mg = wave >> 1, ng = wave & 1;
  const int e0 = blockIdx.x * 128;
#pragma unroll 4
  for (int c = t; c < 128 * 256; c += 256) {
    int row = c >> 8, j = c & 255;
    float xv = ind[e0 + row];
    float h = xv * eW1[j] + eb1[j];
    h = h > 0.f ? h : 0.f;
    smem[row * 264 + j] = (_Float16)h;
  }
  __syncthreads();
  v8f acc[2][4] = {};
  gemm_lds<8, 8>(acc, smem, 264, bbuf, W2s, wave, lane);
  const int n = lane & 15, mofs = (lane >> 4) << 3;
#pragma unroll
  for (int i = 0; i < 2; ++i)
#pragma unroll
    for (int r = 0; r < 8; ++r) {
      int gm = e0 + (mg * 2 + i) * 16 + mofs + r;
#pragma unroll
      for (int j = 0; j < 4; ++j) {
        int col = (ng * 4 + j) * 16 + n;
        float v = acc[i][j][r] + b2[col];
        v = v > 0.f ? v : 0.f;
        emb_out[(size_t)gm * 128 + col] = (_Float16)v;
      }
    }
}

// ---------------------------------------------------------------------------
// Utility kernels
// ---------------------------------------------------------------------------
__global__ void zero_f32(float* p, int n) {
  int i = blockIdx.x * 256 + threadIdx.x;
  if (i < n) p[i] = 0.f;
}
__global__ void count_kernel(const int* __restrict__ pn, const int* __restrict__ cn, float* cp,
                             float* cc, int E) {
  int i = blockIdx.x * 256 + threadIdx.x;
  if (i < E) {
    atomicAdd(cp + pn[i], 1.f);
    atomicAdd(cc + cn[i], 1.f);
  }
}
__global__ void recip_kernel(float* cp, float* cc, int n) {
  int i = blockIdx.x * 256 + threadIdx.x;
  if (i < n) {
    cp[i] = 1.f / fmaxf(cp[i], 1.f);
    cc[i] = 1.f / fmaxf(cc[i], 1.f);
  }
}
// f32 row-major W[K][Nf] -> f16 swizzled B-fragment layout [kk][nt][lane][16]
__global__ void swizzle_kernel(const float* __restrict__ W, _Float16* __restrict__ Ws, int K,
                               int Nf) {
  int id = blockIdx.x * 256 + threadIdx.x;
  if (id >= K * Nf) return;
  int i = id & 15, lane = (id >> 4) & 31, tt = id >> 9;
  int ntiles = Nf >> 4;
  int nt = tt % ntiles, kk = tt / ntiles;
  int k = kk * 32 + ((lane >> 4) << 3) + (i & 7) + ((i >> 3) << 4);
  int n = nt * 16 + (lane & 15);
  Ws[id] = (_Float16)W[(size_t)k * Nf + n];
}

// ---------------------------------------------------------------------------
extern "C" void kernel_launch(void* const* d_in, const int* in_sizes, int n_in, void* d_out,
                              int out_size, void* d_ws, size_t ws_size, hipStream_t stream) {
  const int N = NNODES, E = NEDGES;
  const float* batch_token = (const float*)d_in[0];
  const int* edge_p_node = (const int*)d_in[1];
  const int* edge_c_node = (const int*)d_in[2];
  const float* edge_p_ind = (const float*)d_in[3];
  const float* edge_c_ind = (const float*)d_in[4];
  const float* p_mask = (const float*)d_in[5];
  const float* c_mask = (const float*)d_in[6];
  const float* start_tok = (const float*)d_in[7];
  const float* end_tok = (const float*)d_in[8];
  const float *vW1 = (const float*)d_in[9], *vb1 = (const float*)d_in[10],
              *vW2 = (const float*)d_in[11], *vb2 = (const float*)d_in[12];
  const float *eW1 = (const float*)d_in[13], *eb1 = (const float*)d_in[14],
              *eW2 = (const float*)d_in[15], *eb2 = (const float*)d_in[16];
  const float *pW1 = (const float*)d_in[17], *pb1 = (const float*)d_in[18],
              *pW2 = (const float*)d_in[19], *pb2 = (const float*)d_in[20];
  const float *cW1 = (const float*)d_in[21], *cb1 = (const float*)d_in[22],
              *cW2 = (const float*)d_in[23], *cb2 = (const float*)d_in[24];
  const float *aW1 = (const float*)d_in[25], *ab1 = (const float*)d_in[26],
              *aW2 = (const float*)d_in[27], *ab2 = (const float*)d_in[28];
  float* hidden = (float*)d_out;
  (void)in_sizes; (void)n_in; (void)out_size; (void)ws_size;

  char* ws = (char*)d_ws;
  size_t off = 0;
  auto carve = [&](size_t bytes) -> void* {
    void* p = ws + off;
    off = (off + bytes + 255) & ~(size_t)255;
    return p;
  };
  _Float16* hid_h = (_Float16*)carve((size_t)N * 128 * 2);
  _Float16* epb = (_Float16*)carve((size_t)E * 128 * 2);
  _Float16* ecb = (_Float16*)carve((size_t)E * 128 * 2);
  float* Sp = (float*)carve((size_t)N * 128 * 4);
  float* Sc = (float*)carve((size_t)N * 128 * 4);
  float* cnt_p = (float*)carve((size_t)N * 4);
  float* cnt_c = (float*)carve((size_t)N * 4);
  _Float16* vW1s = (_Float16*)carve(128 * 256 * 2);
  _Float16* vW2s = (_Float16*)carve(256 * 128 * 2);
  _Float16* eW2s = (_Float16*)carve(256 * 128 * 2);
  _Float16* pW1s = (_Float16*)carve(384 * 256 * 2);
  _Float16* pW2s = (_Float16*)carve(256 * 128 * 2);
  _Float16* cW1s = (_Float16*)carve(384 * 256 * 2);
  _Float16* cW2s = (_Float16*)carve(256 * 128 * 2);
  _Float16* aW1s = (_Float16*)carve(384 * 256 * 2);
  _Float16* aW2s = (_Float16*)carve(256 * 128 * 2);

  dim3 b256(256);
  zero_f32<<<(N + 255) / 256, b256, 0, stream>>>(cnt_p, N);
  zero_f32<<<(N + 255) / 256, b256, 0, stream>>>(cnt_c, N);
  count_kernel<<<(E + 255) / 256, b256, 0, stream>>>(edge_p_node, edge_c_node, cnt_p, cnt_c, E);
  recip_kernel<<<(N + 255) / 256, b256, 0, stream>>>(cnt_p, cnt_c, N);
  swizzle_kernel<<<(128 * 256 + 255) / 256, b256, 0, stream>>>(vW1, vW1s, 128, 256);
  swizzle_kernel<<<(256 * 128 + 255) / 256, b256, 0, stream>>>(vW2, vW2s, 256, 128);
  swizzle_kernel<<<(256 * 128 + 255) / 256, b256, 0, stream>>>(eW2, eW2s, 256, 128);
  swizzle_kernel<<<(384 * 256 + 255) / 256, b256, 0, stream>>>(pW1, pW1s, 384, 256);
  swizzle_kernel<<<(256 * 128 + 255) / 256, b256, 0, stream>>>(pW2, pW2s, 256, 128);
  swizzle_kernel<<<(384 * 256 + 255) / 256, b256, 0, stream>>>(cW1, cW1s, 384, 256);
  swizzle_kernel<<<(256 * 128 + 255) / 256, b256, 0, stream>>>(cW2, cW2s, 256, 128);
  swizzle_kernel<<<(384 * 256 + 255) / 256, b256, 0, stream>>>(aW1, aW1s, 384, 256);
  swizzle_kernel<<<(256 * 128 + 255) / 256, b256, 0, stream>>>(aW2, aW2s, 256, 128);
  edge_embed_kernel<<<E / 128, b256, 0, stream>>>(edge_p_ind, eW1, eb1, eW2s, eb2, epb);
  edge_embed_kernel<<<E / 128, b256, 0, stream>>>(edge_c_ind, eW1, eb1, eW2s, eb2, ecb);
  node_init_kernel<<<(N + 127) / 128, b256, 0, stream>>>(batch_token, vW1s, vb1, vW2s, vb2,
                                                         hidden, hid_h);
  for (int hop = 0; hop < 3; ++hop) {
    zero_f32<<<(N * 128 + 255) / 256, b256, 0, stream>>>(Sp, N * 128);
    zero_f32<<<(N * 128 + 255) / 256, b256, 0, stream>>>(Sc, N * 128);
    edge_mlp_kernel<<<E / 128, b256, 0, stream>>>(hid_h, edge_c_node, edge_p_node, epb, pW1s,
                                                  pb1, pW2s, pb2, edge_p_node, Sp);
    edge_mlp_kernel<<<E / 128, b256, 0, stream>>>(hid_h, edge_p_node, edge_c_node, ecb, cW1s,
                                                  cb1, cW2s, cb2, edge_c_node, Sc);
    node_aggr_kernel<<<(N + 127) / 128, b256, 0, stream>>>(
        hid_h, Sp, Sc, cnt_p, cnt_c, p_mask, c_mask, start_tok, end_tok, aW1s, ab1, aW2s, ab2,
        hidden, hid_h);
  }
}
